// MetaGIN_16501264351549
// MI455X (gfx1250) — matrix operand order repaired and checked by reading, not compiled, not run on hardware
//
#include <hip/hip_runtime.h>
#include <cstdint>
#include <cstddef>

typedef _Float16 h16;
typedef __attribute__((ext_vector_type(16))) _Float16 v16h;
typedef __attribute__((ext_vector_type(8)))  float    v8f;

#define TPB 256
#define NNODES 8000
#define NGRAPH 256
#define WID 256
#define EMAXP 45024

static inline dim3 grid1(size_t n) { return dim3((unsigned)((n + TPB - 1) / TPB)); }

// ---------------- elementwise / utility kernels ----------------

__global__ void k_fill(float* __restrict__ p, float v, size_t n) {
  size_t i = (size_t)blockIdx.x * blockDim.x + threadIdx.x;
  if (i < n) p[i] = v;
}

__global__ void k_f32_to_f16_pad(const float* __restrict__ s, h16* __restrict__ d,
                                 int R, int C, int Rp, int Cp) {
  size_t i = (size_t)blockIdx.x * blockDim.x + threadIdx.x;
  size_t n = (size_t)Rp * Cp;
  if (i >= n) return;
  int r = (int)(i / Cp), c = (int)(i % Cp);
  float v = (r < R && c < C) ? s[(size_t)r * C + c] : 0.f;
  d[i] = (h16)v;
}

// ---------------- WMMA GEMM: out[M,N] = X[M,Kp] * W[N,Kp]^T ----------------
// X rows padded to mult-of-32, Kp mult-of-32, N mult-of-32.
// 8 waves/block; each wave computes a 32x32 output block (2x2 WMMA tiles):
// 4 v_wmma per k-step from 2 A + 2 B fragments (2x arithmetic intensity, and
// 4 independent XDL ops in flight to pipeline the matrix unit).
__global__ void k_wmma_gemm(const h16* __restrict__ X, const h16* __restrict__ W,
                            float* __restrict__ out, int M, int N, int Kp) {
  int wave = threadIdx.x >> 5;
  int lane = threadIdx.x & 31;
  int hf = lane >> 4;        // lane half (0: lanes 0-15, 1: lanes 16-31)
  int lr = lane & 15;        // row-in-tile for A, col-in-tile for B/C
  int m0 = (blockIdx.y * 8 + wave) * 32;
  int Mp = (M + 31) & ~31;
  if (m0 >= Mp) return;      // uniform per wave: EXEC stays all-ones
  int n0 = blockIdx.x * 32;
  const h16* Xr0 = X + (size_t)(m0 + lr) * Kp;
  const h16* Xr1 = Xr0 + (size_t)16 * Kp;
  const h16* Wr0 = W + (size_t)(n0 + lr) * Kp;
  const h16* Wr1 = Wr0 + (size_t)16 * Kp;
  v8f acc00 = {}, acc01 = {}, acc10 = {}, acc11 = {};
  for (int k0 = 0; k0 < Kp; k0 += 32) {
    v16h a0, a1, b0, b1;
#pragma unroll
    for (int v = 0; v < 8; ++v) {
      // A 16x32 f16 layout (ISA 7.12.2): vgpr v: K = (v>=4)*16 + 2*(v&3) + 8*half
      int ka = ((v >> 2) << 4) + ((v & 3) << 1) + (hf << 3);
      a0[2 * v]     = Xr0[k0 + ka];
      a0[2 * v + 1] = Xr0[k0 + ka + 1];
      a1[2 * v]     = Xr1[k0 + ka];
      a1[2 * v + 1] = Xr1[k0 + ka + 1];
      // B 32x16 f16: same K order as A: vgpr v: K = (v>=4)*16 + 2*(v&3) + 8*half
      int kb = ((v >> 2) << 4) + ((v & 3) << 1) + (hf << 3);
      b0[2 * v]     = Wr0[k0 + kb];
      b0[2 * v + 1] = Wr0[k0 + kb + 1];
      b1[2 * v]     = Wr1[k0 + kb];
      b1[2 * v + 1] = Wr1[k0 + kb + 1];
    }
    acc00 = __builtin_amdgcn_wmma_f32_16x16x32_f16(false, a0, false, b0, (short)0, acc00, false, false);
    acc01 = __builtin_amdgcn_wmma_f32_16x16x32_f16(false, a0, false, b1, (short)0, acc01, false, false);
    acc10 = __builtin_amdgcn_wmma_f32_16x16x32_f16(false, a1, false, b0, (short)0, acc10, false, false);
    acc11 = __builtin_amdgcn_wmma_f32_16x16x32_f16(false, a1, false, b1, (short)0, acc11, false, false);
  }
  // C/D layout: vgpr v -> row v (lanes 0-15) / 8+v (lanes 16-31); col = lane%16
#pragma unroll
  for (int v = 0; v < 8; ++v) {
    int mr = v + (hf << 3);
    int m0r = m0 + mr, m1r = m0 + 16 + mr;
    if (m0r < M) {
      out[(size_t)m0r * N + n0 + lr]      = acc00[v];
      out[(size_t)m0r * N + n0 + 16 + lr] = acc01[v];
    }
    if (m1r < M) {
      out[(size_t)m1r * N + n0 + lr]      = acc10[v];
      out[(size_t)m1r * N + n0 + 16 + lr] = acc11[v];
    }
  }
}

// ---------------- GroupNorm (group size always 16), in place ----------------
__global__ void k_groupnorm(float* __restrict__ x, int rows, int wn) {
  int groups = wn >> 4;
  size_t i = (size_t)blockIdx.x * blockDim.x + threadIdx.x;
  if (i >= (size_t)rows * groups) return;
  size_t base = (i / groups) * (size_t)wn + (i % groups) * 16;
  float m = 0.f;
#pragma unroll
  for (int j = 0; j < 16; ++j) m += x[base + j];
  m *= (1.f / 16.f);
  float v = 0.f;
#pragma unroll
  for (int j = 0; j < 16; ++j) { float d = x[base + j] - m; v += d * d; }
  v *= (1.f / 16.f);
  float r = rsqrtf(v + 1e-5f);
#pragma unroll
  for (int j = 0; j < 16; ++j) x[base + j] = (x[base + j] - m) * r;
}

// ---------------- grouped 1x1 conv gate/value + relu*mul, optional emb bias ----
__global__ void k_grouped(const float* __restrict__ xn, const float* __restrict__ gate,
                          const float* __restrict__ value, float* __restrict__ out,
                          int rows, int wn, int nh, int dout,
                          const float* __restrict__ emb, const int* __restrict__ attr) {
  int wa = nh * dout;
  size_t i = (size_t)blockIdx.x * blockDim.x + threadIdx.x;
  if (i >= (size_t)rows * wa) return;
  int row = (int)(i / wa);
  int o = (int)(i % wa);
  int h = o / dout, oo = o % dout;
  const float* xr = xn + (size_t)row * wn + h * 16;
  const float* gr = gate + (size_t)(h * dout + oo) * 16;
  const float* vr = value + (size_t)(h * dout + oo) * 16;
  const float* er = emb ? emb + (size_t)attr[row] * wn + h * 16 : nullptr;
  float g = 0.f, vv = 0.f;
#pragma unroll
  for (int j = 0; j < 16; ++j) {
    float xi = xr[j];
    float xg = er ? xi + er[j] : xi;
    g += xg * gr[j];
    vv += xi * vr[j];
  }
  out[i] = fmaxf(g, 0.f) * vv;
}

// ---------------- graph-specific kernels ----------------

__global__ void k_embbag3(const int* __restrict__ xi, const float* __restrict__ emb,
                          float* __restrict__ out) {
  size_t i = (size_t)blockIdx.x * blockDim.x + threadIdx.x;
  if (i >= (size_t)NNODES * WID) return;
  int n = (int)(i / WID), c = (int)(i % WID);
  float v = emb[(size_t)xi[n * 3 + 0] * WID + c] +
            emb[(size_t)xi[n * 3 + 1] * WID + c] +
            emb[(size_t)xi[n * 3 + 2] * WID + c];
  out[i] = v * (1.f / 3.f);
}

__global__ void k_edge_assemble(const float* __restrict__ xs, const float* __restrict__ xt,
                                const int* __restrict__ eidx, int E, float* __restrict__ out) {
  size_t i = (size_t)blockIdx.x * blockDim.x + threadIdx.x;
  if (i >= (size_t)E * 512) return;
  int e = (int)(i / 512), c = (int)(i % 512);
  int s = eidx[e], t = eidx[E + e];
  out[i] = xs[(size_t)s * 512 + c] + xt[(size_t)t * 512 + c];
}

__global__ void k_scatter_add(const float* __restrict__ src, const int* __restrict__ eidx,
                              int E, float* __restrict__ acc) {
  size_t i = (size_t)blockIdx.x * blockDim.x + threadIdx.x;
  if (i >= (size_t)E * WID) return;
  int e = (int)(i / WID), c = (int)(i % WID);
  atomicAdd(&acc[(size_t)eidx[E + e] * WID + c], src[i]);
}

__global__ void k_deg_count(const int* __restrict__ eidx, int E, float* __restrict__ deg) {
  size_t i = (size_t)blockIdx.x * blockDim.x + threadIdx.x;
  if (i < (size_t)E) atomicAdd(&deg[eidx[E + i]], 1.f);
}

__global__ void k_log_max1(float* __restrict__ d, int n) {
  size_t i = (size_t)blockIdx.x * blockDim.x + threadIdx.x;
  if (i < (size_t)n) d[i] = logf(fmaxf(d[i], 1.f));
}

__global__ void k_hop_scale(const float* __restrict__ acc, const float* __restrict__ ldeg,
                            const float* __restrict__ degp, float* __restrict__ x_hop,
                            float* __restrict__ x_out) {
  size_t i = (size_t)blockIdx.x * blockDim.x + threadIdx.x;
  if (i >= (size_t)NNODES * WID) return;
  int n = (int)(i / WID), c = (int)(i % WID);
  float v = expf(degp[c] * ldeg[n]) * acc[i];
  x_hop[i] = v;
  x_out[i] += v;
}

__global__ void k_scale_exp_add(const float* __restrict__ x, const float* __restrict__ res,
                                const float* __restrict__ s, float* __restrict__ out, size_t n) {
  size_t i = (size_t)blockIdx.x * blockDim.x + threadIdx.x;
  if (i >= n) return;
  out[i] = expf(s[i % WID]) * x[i] + res[i];
}

__global__ void k_seg_sum(const float* __restrict__ x, const int* __restrict__ batch,
                          float* __restrict__ out, int rows) {
  size_t i = (size_t)blockIdx.x * blockDim.x + threadIdx.x;
  if (i >= (size_t)rows * WID) return;
  int n = (int)(i / WID), c = (int)(i % WID);
  atomicAdd(&out[(size_t)batch[n] * WID + c], x[i]);
}

__global__ void k_cnt(const int* __restrict__ batch, float* __restrict__ cnt) {
  size_t i = (size_t)blockIdx.x * blockDim.x + threadIdx.x;
  if (i < (size_t)NNODES) atomicAdd(&cnt[batch[i]], 1.f);
}

__global__ void k_virt_gather_add(const float* __restrict__ vout, const int* __restrict__ batch,
                                  float* __restrict__ h_out) {
  size_t i = (size_t)blockIdx.x * blockDim.x + threadIdx.x;
  if (i >= (size_t)NNODES * WID) return;
  int n = (int)(i / WID), c = (int)(i % WID);
  h_out[i] += vout[(size_t)batch[n] * WID + c];
}

__global__ void k_combine(const float* __restrict__ x0, const float* __restrict__ x1s,
                          const float* __restrict__ cnt, float* __restrict__ out) {
  size_t i = (size_t)blockIdx.x * blockDim.x + threadIdx.x;
  if (i >= (size_t)NGRAPH * WID) return;
  int g = (int)(i / WID);
  out[i] = x0[i] + x1s[i] / fmaxf(cnt[g], 1.f);
}

__global__ void k_head_final(const float* __restrict__ xx, const float* __restrict__ post,
                             float* __restrict__ out) {
  int g = blockIdx.x * blockDim.x + threadIdx.x;
  if (g >= NGRAPH) return;
  float s = 0.f;
  for (int c = 0; c < 512; ++c) s += xx[(size_t)g * 512 + c] * post[c];
  out[g] = (s + 1.f) * 5.5f;
}

// ---------------- host-side param plumbing ----------------

struct BlockW { const float *deg, *emb, *gate, *post, *value, *src, *tgt; };
struct GlbW   { const float *gate, *post, *pre, *value; };
struct MfW    { GlbW ffn; const float *sca_post, *sca_pre; };
struct BlockH { const h16 *src, *tgt, *post; };
struct GlbH   { const h16 *pre, *post; };

struct PCur {
  void* const* din; int nin; const float* packed; size_t off; int li;
  const float* next(size_t cnt) {
    if (nin > 10) return (const float*)din[9 + li++];
    const float* p = packed + off; off += cnt; return p;
  }
};

extern "C" void kernel_launch(void* const* d_in, const int* in_sizes, int n_in,
                              void* d_out, int out_size, void* d_ws, size_t ws_size,
                              hipStream_t stream) {
  (void)in_sizes; (void)out_size; (void)ws_size;
  const int*   xi    = (const int*)d_in[0];
  const float* z     = (const float*)d_in[1];
  const int*   ei[3] = {(const int*)d_in[2], (const int*)d_in[4], (const int*)d_in[6]};
  const int*   ea[3] = {(const int*)d_in[3], (const int*)d_in[5], (const int*)d_in[7]};
  const int*   batch = (const int*)d_in[8];
  const int EE[3]  = {15000, 30000, 45000};
  const int EBv[3] = {33, 3, 4};

  // -------- walk params (jax pytree flatten order: sorted dict keys) --------
  PCur pc{d_in, n_in, (n_in == 10) ? (const float*)d_in[9] : nullptr, 0, 0};
  auto read_block = [&](int hop) {
    BlockW b;
    b.deg   = pc.next(256);
    b.emb   = pc.next((size_t)EBv[hop] * 512);
    b.gate  = pc.next(32 * 16 * 16);
    b.post  = pc.next(256 * 512);
    b.value = pc.next(32 * 16 * 16);
    b.src   = pc.next(512 * 256);
    b.tgt   = pc.next(512 * 256);
    return b;
  };
  auto read_glb = [&](size_t gn_, size_t pn, size_t prn) {
    GlbW g;
    g.gate = pc.next(gn_); g.post = pc.next(pn); g.pre = pc.next(prn); g.value = pc.next(gn_);
    return g;
  };
  auto read_mf = [&](bool resid) {
    MfW m; m.ffn = read_glb(16 * 32 * 16, 256 * 512, 256 * 256);
    m.sca_post = resid ? pc.next(256) : nullptr;
    m.sca_pre  = pc.next(256);
    return m;
  };

  BlockW atomconv[3];
  for (int h = 0; h < 3; ++h) atomconv[h] = read_block(h);
  const float* atom_emb = pc.next(138 * 256);
  MfW  atom_main = read_mf(false);
  GlbW atom_pos  = read_glb(16 * 32 * 16, 256 * 512, 256 * 16);
  BlockW convb[2][6];
  for (int l = 0; l < 2; ++l)
    for (int i = 0; i < 6; ++i) convb[l][i] = read_block(i % 3);
  GlbW head_head = read_glb(32 * 16 * 16, 512, 512 * 256);
  GlbW head_node = read_glb(16 * 32 * 16, 256 * 512, 256 * 256);
  MfW  head_virt = read_mf(false);
  MfW  main0 = read_mf(false), main1 = read_mf(true);
  GlbW virt0 = read_glb(16 * 32 * 16, 256 * 512, 256 * 256);
  GlbW virt1 = read_glb(16 * 32 * 16, 256 * 512, 256 * 256);

  // -------- workspace arena --------
  uint8_t* wsb = (uint8_t*)d_ws; size_t cur = 0;
  auto alloc  = [&](size_t bytes) { cur = (cur + 255) & ~(size_t)255; void* p = wsb + cur; cur += bytes; return p; };
  auto allocF = [&](size_t n) { return (float*)alloc(n * 4); };
  auto allocH = [&](size_t n) { return (h16*)alloc(n * 2); };

  float* nodeA = allocF((size_t)NNODES * 256);   // h_in
  float* nodeB = allocF((size_t)NNODES * 256);   // h_out
  float* nodeC = allocF((size_t)NNODES * 256);   // x_hop
  float* nodeD = allocF((size_t)NNODES * 256);   // acc / xx / x1
  float* nodeE = allocF((size_t)NNODES * 256);   // xn scratch
  float* n512a = allocF((size_t)NNODES * 512);   // xs / grouped-out
  float* n512b = allocF((size_t)NNODES * 512);   // xt
  float* edgeA = allocF((size_t)EMAXP * 512);    // edge xn (GN in-place) / outE alias
  float* edgeB = allocF((size_t)EMAXP * 512);    // edge grouped-out
  float* gA = allocF((size_t)NGRAPH * 256);      // h_virt
  float* gB = allocF((size_t)NGRAPH * 256);
  float* gC = allocF((size_t)NGRAPH * 256);
  float* gF = allocF((size_t)NGRAPH * 256);
  float* gD = allocF((size_t)NGRAPH * 512);
  float* gE = allocF((size_t)NGRAPH * 512);
  float* ldeg = allocF((size_t)3 * NNODES);
  float* cnt  = allocF(NGRAPH);
  h16* hZ    = allocH((size_t)NNODES * 32);
  h16* hN256 = allocH((size_t)NNODES * 256);
  h16* hN512 = allocH((size_t)NNODES * 512);
  h16* hEdge = allocH((size_t)EMAXP * 512);
  h16* hG    = allocH((size_t)NGRAPH * 512);

  // -------- launch helpers --------
  auto fill = [&](float* p, float v, size_t n) {
    k_fill<<<grid1(n), TPB, 0, stream>>>(p, v, n);
  };
  auto cvt = [&](const float* s, h16* d, int R, int C, int Rp, int Cp) {
    k_f32_to_f16_pad<<<grid1((size_t)Rp * Cp), TPB, 0, stream>>>(s, d, R, C, Rp, Cp);
  };
  auto gemm = [&](const h16* X, const h16* W, float* out, int M, int N, int Kp) {
    int Mp = (M + 31) & ~31;
    dim3 g(N / 32, (Mp + 255) / 256);
    k_wmma_gemm<<<g, dim3(TPB), 0, stream>>>(X, W, out, M, N, Kp);
  };
  auto gnorm = [&](float* x, int rows, int wn) {
    k_groupnorm<<<grid1((size_t)rows * (wn >> 4)), TPB, 0, stream>>>(x, rows, wn);
  };
  auto grouped = [&](const float* xn, const float* gate, const float* value, float* out,
                     int rows, int wn, int nh, int dout, const float* emb, const int* attr) {
    k_grouped<<<grid1((size_t)rows * nh * dout), TPB, 0, stream>>>(xn, gate, value, out,
                                                                   rows, wn, nh, dout, emb, attr);
  };
  auto halfw = [&](const float* w, int R, int C, int Cp) {
    h16* d = allocH((size_t)R * Cp);
    cvt(w, d, R, C, R, Cp);
    return d;
  };

  // -------- f16 weight conversions --------
  BlockH atomconvH[3], convbH[2][6];
  for (int h = 0; h < 3; ++h)
    atomconvH[h] = {halfw(atomconv[h].src, 512, 256, 256), halfw(atomconv[h].tgt, 512, 256, 256),
                    halfw(atomconv[h].post, 256, 512, 512)};
  for (int l = 0; l < 2; ++l)
    for (int i = 0; i < 6; ++i)
      convbH[l][i] = {halfw(convb[l][i].src, 512, 256, 256), halfw(convb[l][i].tgt, 512, 256, 256),
                      halfw(convb[l][i].post, 256, 512, 512)};
  GlbH atom_posH  = {halfw(atom_pos.pre, 256, 16, 32),  halfw(atom_pos.post, 256, 512, 512)};
  GlbH atom_mainH = {halfw(atom_main.ffn.pre, 256, 256, 256), halfw(atom_main.ffn.post, 256, 512, 512)};
  GlbH main0H     = {halfw(main0.ffn.pre, 256, 256, 256), halfw(main0.ffn.post, 256, 512, 512)};
  GlbH main1H     = {halfw(main1.ffn.pre, 256, 256, 256), halfw(main1.ffn.post, 256, 512, 512)};
  GlbH virt0H     = {halfw(virt0.pre, 256, 256, 256),     halfw(virt0.post, 256, 512, 512)};
  GlbH virt1H     = {halfw(virt1.pre, 256, 256, 256),     halfw(virt1.post, 256, 512, 512)};
  GlbH head_virtH = {halfw(head_virt.ffn.pre, 256, 256, 256), halfw(head_virt.ffn.post, 256, 512, 512)};
  GlbH head_nodeH = {halfw(head_node.pre, 256, 256, 256), halfw(head_node.post, 256, 512, 512)};
  const h16* head_head_preH = halfw(head_head.pre, 512, 256, 256);

  // node GLB (pre 256xK -> GN -> grouped -> post): out fp32 [rows,256]
  auto glb_node = [&](int rows, const float* xin, const GlbW& w, const GlbH& wh,
                      int Kin, int Kp, h16* xin_h, float* xn, float* xx, h16* xx_h,
                      float* out, int wn, int nh, int dout) {
    cvt(xin, xin_h, rows, Kin, rows, Kp);
    gemm(xin_h, wh.pre, xn, rows, wn, Kp);
    gnorm(xn, rows, wn);
    grouped(xn, w.gate, w.value, xx, rows, wn, nh, dout, nullptr, nullptr);
    int wa = nh * dout;
    cvt(xx, xx_h, rows, wa, rows, wa);
    gemm(xx_h, wh.post, out, rows, 256, wa);
  };

  auto conv_block = [&](const BlockW& w, const BlockH& wh, int hop,
                        float* x_hop, float* x_out) {
    int E = EE[hop], Ep = (E + 31) & ~31;
    cvt(x_hop, hN256, NNODES, 256, NNODES, 256);
    gemm(hN256, wh.src, n512a, NNODES, 512, 256);                       // xs
    gemm(hN256, wh.tgt, n512b, NNODES, 512, 256);                       // xt
    k_edge_assemble<<<grid1((size_t)E * 512), TPB, 0, stream>>>(n512a, n512b, ei[hop], E, edgeA);
    gnorm(edgeA, E, 512);
    grouped(edgeA, w.gate, w.value, edgeB, E, 512, 32, 16, w.emb, ea[hop]);
    cvt(edgeB, hEdge, E, 512, Ep, 512);
    gemm(hEdge, wh.post, edgeA, E, 256, 512);                           // outE (alias)
    fill(nodeD, 0.f, (size_t)NNODES * 256);
    k_scatter_add<<<grid1((size_t)E * 256), TPB, 0, stream>>>(edgeA, ei[hop], E, nodeD);
    k_hop_scale<<<grid1((size_t)NNODES * 256), TPB, 0, stream>>>(nodeD, ldeg + (size_t)hop * NNODES,
                                                                 w.deg, x_hop, x_out);
  };

  // ================= forward pass =================
  // degrees (log), graph counts
  for (int h = 0; h < 3; ++h) {
    fill(ldeg + (size_t)h * NNODES, 0.f, NNODES);
    k_deg_count<<<grid1(EE[h]), TPB, 0, stream>>>(ei[h], EE[h], ldeg + (size_t)h * NNODES);
    k_log_max1<<<grid1(NNODES), TPB, 0, stream>>>(ldeg + (size_t)h * NNODES, NNODES);
  }
  fill(cnt, 0.f, NGRAPH);
  k_cnt<<<grid1(NNODES), TPB, 0, stream>>>(batch, cnt);

  // h_in = embedding bag; h_out = GLB(atom_pos, z)
  k_embbag3<<<grid1((size_t)NNODES * 256), TPB, 0, stream>>>(xi, atom_emb, nodeA);
  glb_node(NNODES, z, atom_pos, atom_posH, 16, 32, hZ, nodeE, n512a, hN512, nodeB, 256, 16, 32);

  // atom_conv: x_hop = h_in, x_out = h_out (accumulate)
  hipMemcpyAsync(nodeC, nodeA, (size_t)NNODES * 256 * 4, hipMemcpyDeviceToDevice, stream);
  for (int i = 0; i < 3; ++i) conv_block(atomconv[i], atomconvH[i], i, nodeC, nodeB);

  // atom_main (no residual): h_in = GLB(ffn, exp(sca_pre)*h_in + h_out)
  k_scale_exp_add<<<grid1((size_t)NNODES * 256), TPB, 0, stream>>>(nodeA, nodeB, atom_main.sca_pre,
                                                                   nodeD, (size_t)NNODES * 256);
  glb_node(NNODES, nodeD, atom_main.ffn, atom_mainH, 256, 256, hN256, nodeE, n512a, hN512,
           nodeA, 256, 16, 32);

  // depth loop
  fill(gA, 0.f, (size_t)NGRAPH * 256);  // h_virt
  for (int l = 0; l < 2; ++l) {
    fill(nodeB, 0.f, (size_t)NNODES * 256);
    hipMemcpyAsync(nodeC, nodeA, (size_t)NNODES * 256 * 4, hipMemcpyDeviceToDevice, stream);
    for (int i = 0; i < 6; ++i) conv_block(convb[l][i], convbH[l][i], i % 3, nodeC, nodeB);
    k_seg_sum<<<grid1((size_t)NNODES * 256), TPB, 0, stream>>>(nodeA, batch, gA, NNODES);
    const GlbW& vw = (l == 0) ? virt0 : virt1;
    const GlbH& vh = (l == 0) ? virt0H : virt1H;
    glb_node(NGRAPH, gA, vw, vh, 256, 256, hG, gC, gD, hG, gB, 256, 16, 32);
    k_virt_gather_add<<<grid1((size_t)NNODES * 256), TPB, 0, stream>>>(gB, batch, nodeB);
    const MfW& mw = (l == 0) ? main0 : main1;
    const GlbH& mh = (l == 0) ? main0H : main1H;
    k_scale_exp_add<<<grid1((size_t)NNODES * 256), TPB, 0, stream>>>(nodeA, nodeB, mw.sca_pre,
                                                                     nodeD, (size_t)NNODES * 256);
    if (l == 0) {
      glb_node(NNODES, nodeD, mw.ffn, mh, 256, 256, hN256, nodeE, n512a, hN512, nodeA, 256, 16, 32);
    } else {
      glb_node(NNODES, nodeD, mw.ffn, mh, 256, 256, hN256, nodeE, n512a, hN512, nodeC, 256, 16, 32);
      k_scale_exp_add<<<grid1((size_t)NNODES * 256), TPB, 0, stream>>>(nodeD, nodeC, mw.sca_post,
                                                                       nodeA, (size_t)NNODES * 256);
    }
  }

  // head
  fill(gB, 0.f, (size_t)NGRAPH * 256);
  k_seg_sum<<<grid1((size_t)NNODES * 256), TPB, 0, stream>>>(nodeA, batch, gB, NNODES);   // x0
  k_scale_exp_add<<<grid1((size_t)NGRAPH * 256), TPB, 0, stream>>>(gA, gB, head_virt.sca_pre,
                                                                   gC, (size_t)NGRAPH * 256);
  glb_node(NGRAPH, gC, head_virt.ffn, head_virtH, 256, 256, hG, gB, gD, hG, gC, 256, 16, 32); // x0'
  glb_node(NNODES, nodeA, head_node, head_nodeH, 256, 256, hN256, nodeE, n512a, hN512,
           nodeD, 256, 16, 32);                                                             // x1
  fill(gF, 0.f, (size_t)NGRAPH * 256);
  k_seg_sum<<<grid1((size_t)NNODES * 256), TPB, 0, stream>>>(nodeD, batch, gF, NNODES);
  k_combine<<<grid1((size_t)NGRAPH * 256), TPB, 0, stream>>>(gC, gF, cnt, gB);              // s

  // head.head GLB: pre [512,256] -> GN(32) -> grouped(32,16) -> dot with post[1,512]
  cvt(gB, hG, NGRAPH, 256, NGRAPH, 256);
  gemm(hG, head_head_preH, gD, NGRAPH, 512, 256);
  gnorm(gD, NGRAPH, 512);
  grouped(gD, head_head.gate, head_head.value, gE, NGRAPH, 512, 32, 16, nullptr, nullptr);
  k_head_final<<<grid1(NGRAPH), TPB, 0, stream>>>(gE, head_head.post, (float*)d_out);
}
